// BI_Mamba_12360915878184
// MI455X (gfx1250) — matrix operand order repaired and checked by reading, not compiled, not run on hardware
//
#include <hip/hip_runtime.h>
#include <hip/hip_bf16.h>
#include <math.h>

// ---------------------------------------------------------------------------
// BI-Mamba on gfx1250: WMMA bf16 GEMMs + async-LDS staging + VALU conv/scan
// ---------------------------------------------------------------------------
#define BB   16
#define LL   512
#define DM   256         // D_MODEL
#define DI   512         // D_INNER
#define NS   16          // D_STATE
#define RK   16          // DT_RANK
#define RTOT (BB * LL)   // 8192 rows

typedef __bf16 bf16_t;
typedef __attribute__((ext_vector_type(16))) __bf16 v16bf;
typedef __attribute__((ext_vector_type(8)))  __bf16 v8bf;
typedef __attribute__((ext_vector_type(8)))  float  v8f;

// ----------------------- CDNA5 async global->LDS helpers --------------------
// LDS operand is the wave-relative LDS byte offset (addr[31:0] per aperture).
__device__ __forceinline__ void async_copy_b128(const void* g, void* lds) {
  unsigned l = (unsigned)(size_t)lds;
  asm volatile("global_load_async_to_lds_b128 %0, %1, off"
               :: "v"(l), "v"(g) : "memory");
}
__device__ __forceinline__ void async_copy_b32(const void* g, void* lds) {
  unsigned l = (unsigned)(size_t)lds;
  asm volatile("global_load_async_to_lds_b32 %0, %1, off"
               :: "v"(l), "v"(g) : "memory");
}
__device__ __forceinline__ void wait_async0() {
  asm volatile("s_wait_asynccnt 0x0" ::: "memory");
}

// ----------------------- WMMA fragment loaders -----------------------------
// A fragment: 16x32 (MxK) bf16, row-major source with leading dim `lda`.
// ISA 7.12.2: lane<16 -> M=lane, elems 0..7 = K 0..7, elems 8..15 = K 16..23
//             lane>=16 -> M=lane-16, K 8..15 and K 24..31.
__device__ __forceinline__ v16bf load_a_frag(const bf16_t* __restrict__ tile,
                                             int lda, int lane) {
  const int m  = lane & 15;
  const int kb = (lane >> 4) << 3;               // 0 or 8
  const v8bf lo = *(const v8bf*)(tile + (size_t)m * lda + kb);
  const v8bf hi = *(const v8bf*)(tile + (size_t)m * lda + 16 + kb);
  v16bf r;
#pragma unroll
  for (int i = 0; i < 8; ++i) { r[i] = lo[i]; r[i + 8] = hi[i]; }
  return r;
}

// B fragment from pre-packed weights: one contiguous 32B load per lane.
// Packed layout: ((tn*KT + tk)*32 + lane)*16 + e
__device__ __forceinline__ v16bf load_b_frag(const bf16_t* __restrict__ packed,
                                             int tn, int tk, int KT, int lane) {
  const bf16_t* p = packed + (((size_t)tn * KT + tk) * 32 + lane) * 16;
  return *(const v16bf*)p;
}

__device__ __forceinline__ v8f wmma_bf16(v16bf a, v16bf b, v8f c) {
  return __builtin_amdgcn_wmma_f32_16x16x32_bf16(false, a, false, b,
                                                 (short)0, c, false, false);
}

// ----------------------- weight packer -------------------------------------
// W: (K,N) row-major fp32 -> packed bf16 B-fragments.
__global__ void pack_w_kernel(const float* __restrict__ W, bf16_t* __restrict__ out,
                              int K, int N) {
  int idx = blockIdx.x * 256 + threadIdx.x;
  if (idx >= K * N) return;
  int e    = idx & 15;
  int lane = (idx >> 4) & 31;
  int tile = idx >> 9;
  int KT   = K >> 5;
  int tn   = tile / KT;
  int tk   = tile - tn * KT;
  int n    = (tn << 4) + (lane & 15);
  int k    = (tk << 5) + ((lane >> 4) << 3) + ((e < 8) ? e : (8 + e));
  out[idx] = (bf16_t)W[(size_t)k * N + n];
}

// ----------------------- RMSNorm (+optional L-flip) -> bf16 -----------------
__global__ void rmsnorm_kernel(const float* __restrict__ x, const float* __restrict__ w,
                               bf16_t* __restrict__ out, int flip) {
  __shared__ float red[DM];
  const int r = blockIdx.x;              // destination row (flipped domain)
  const int b = r >> 9, l = r & (LL - 1);
  const int sl = flip ? (LL - 1 - l) : l;
  const size_t sr = ((size_t)b << 9) + sl;
  const int t = threadIdx.x;
  float v = x[sr * DM + t];
  red[t] = v * v;
  __syncthreads();
  for (int s = 128; s > 0; s >>= 1) {
    if (t < s) red[t] += red[t + s];
    __syncthreads();
  }
  float scale = rsqrtf(red[0] * (1.0f / DM) + 1e-5f);
  out[(size_t)r * DM + t] = (bf16_t)(v * scale * w[t]);
}

// ----------------------- GEMM1: xn(8192x256) @ in_w(256x1024) ---------------
// grid (512, 2), block 256. A panel (16x256, 8KB) staged into LDS via async
// b128 copies and shared by all 8 waves; each wave: 16x64 out (4 accum).
__global__ void gemm_xz_kernel(const bf16_t* __restrict__ xn,
                               const bf16_t* __restrict__ wpak,
                               float* __restrict__ xm, float* __restrict__ z) {
  __shared__ __align__(16) bf16_t As[16 * DM];   // 8KB
  const int tid  = threadIdx.x;
  const int lane = tid & 31;
  const int wave = tid >> 5;
  const int mt   = blockIdx.x;
  const int ntb  = blockIdx.y * 32 + wave * 4;   // 16-col tile index
  const int KT   = DM / 32;                      // 8

  const bf16_t* asrc = xn + (size_t)mt * 16 * DM;
#pragma unroll
  for (int rnd = 0; rnd < 2; ++rnd) {            // 2 * 256 thr * 16B = 8KB
    int byteOff = (rnd * 256 + tid) * 16;
    async_copy_b128((const char*)asrc + byteOff, (char*)As + byteOff);
  }
  wait_async0();
  __syncthreads();

  v8f a0 = {}, a1 = {}, a2 = {}, a3 = {};
  for (int tk = 0; tk < KT; ++tk) {
    if (tk + 1 < KT)
      __builtin_prefetch(wpak + (((size_t)ntb * KT + tk + 1) * 32 + lane) * 16, 0, 1);
    v16bf a  = load_a_frag(As + tk * 32, DM, lane);      // ds_load
    v16bf b0 = load_b_frag(wpak, ntb + 0, tk, KT, lane);
    v16bf b1 = load_b_frag(wpak, ntb + 1, tk, KT, lane);
    v16bf b2 = load_b_frag(wpak, ntb + 2, tk, KT, lane);
    v16bf b3 = load_b_frag(wpak, ntb + 3, tk, KT, lane);
    a0 = wmma_bf16(a, b0, a0);
    a1 = wmma_bf16(a, b1, a1);
    a2 = wmma_bf16(a, b2, a2);
    a3 = wmma_bf16(a, b3, a3);
  }
  const int col0 = ntb * 16 + (lane & 15);
  const int rb   = mt * 16 + ((lane >> 4) << 3);
  v8f acc[4] = {a0, a1, a2, a3};
#pragma unroll
  for (int j = 0; j < 4; ++j) {
    int c = col0 + j * 16;
    float* dst = (c < DI) ? xm : z;
    int cc = (c < DI) ? c : (c - DI);
#pragma unroll
    for (int i = 0; i < 8; ++i)
      dst[(size_t)(rb + i) * DI + cc] = acc[j][i];
  }
}

// ----------------------- causal depthwise conv + SiLU -----------------------
__global__ void conv_silu_kernel(const float* __restrict__ xm,
                                 const float* __restrict__ cw,
                                 const float* __restrict__ cb,
                                 float* __restrict__ xa) {
  size_t idx = (size_t)blockIdx.x * 256 + threadIdx.x;   // RTOT*DI threads
  int d = idx & (DI - 1);
  size_t r = idx >> 9;
  int l = (int)(r & (LL - 1));
  float acc = cb[d];
#pragma unroll
  for (int j = 0; j < 4; ++j) {
    int ll = l - 3 + j;
    if (ll >= 0) acc += xm[(r - l + ll) * DI + d] * cw[j * DI + d];
  }
  xa[idx] = acc / (1.0f + __expf(-acc));
}

// ----------------------- x-proj + delta-proj + softplus ---------------------
__global__ void dbc_delta_kernel(const float* __restrict__ xa,
                                 const float* __restrict__ xproj,
                                 const float* __restrict__ dtw,
                                 const float* __restrict__ dtb,
                                 float* __restrict__ delta,
                                 float* __restrict__ Bm, float* __restrict__ Cm) {
  __shared__ float xrow[DI];
  __shared__ float dbc[RK + 2 * NS];
  const int r = blockIdx.x, t = threadIdx.x;             // 256 threads
  xrow[t]       = xa[(size_t)r * DI + t];
  xrow[t + 256] = xa[(size_t)r * DI + t + 256];
  __syncthreads();
  if (t < RK + 2 * NS) {
    float s = 0.f;
    for (int k = 0; k < DI; ++k) s += xrow[k] * xproj[(size_t)k * 48 + t];
    dbc[t] = s;
    if (t >= RK && t < RK + NS)       Bm[(size_t)r * NS + (t - RK)] = s;
    else if (t >= RK + NS)            Cm[(size_t)r * NS + (t - RK - NS)] = s;
  }
  __syncthreads();
#pragma unroll
  for (int h = 0; h < 2; ++h) {
    int d = t + h * 256;
    float s = dtb[d];
#pragma unroll
    for (int k = 0; k < RK; ++k) s += dbc[k] * dtw[(size_t)k * DI + d];
    delta[(size_t)r * DI + d] = (s > 20.f) ? s : log1pf(__expf(s));
  }
}

// ----------------------- sequential selective scan --------------------------
// grid 64 = B * (DI/128); block 128. Lane owns one (b,d) channel, 16-state in
// regs. B/C vectors double-buffered in LDS via async b32 loads so step l+1's
// 128B fly in behind step l's exp/FMA chain; one barrier per step.
__global__ void scan_kernel(const float* __restrict__ delta,
                            const float* __restrict__ xa,
                            const float* __restrict__ Bm,
                            const float* __restrict__ Cm,
                            const float* __restrict__ Alog,
                            const float* __restrict__ Dp,
                            float* __restrict__ y) {
  __shared__ __align__(16) float BC[2][2 * NS];   // [buf][B:0..15 | C:16..31]
  const int b = blockIdx.x >> 2;
  const int d = ((blockIdx.x & 3) << 7) + threadIdx.x;
  float A[NS], h[NS];
#pragma unroll
  for (int n = 0; n < NS; ++n) {
    A[n] = -__expf(Alog[(size_t)d * NS + n]);
    h[n] = 0.f;
  }
  const float dpd = Dp[d];

  // wave0 issues the async BC stage for step l into buffer `buf`
  auto issue = [&](int l, int buf) {
    if (threadIdx.x < 2 * NS) {
      size_t r = ((size_t)b << 9) + l;
      const float* g = (threadIdx.x < NS)
                           ? (Bm + r * NS + threadIdx.x)
                           : (Cm + r * NS + (threadIdx.x - NS));
      async_copy_b32(g, &BC[buf][threadIdx.x]);
    }
  };

  issue(0, 0);
  int cur = 0;
  for (int l = 0; l < LL; ++l) {
    wait_async0();          // wave0 waits for its ASYNCcnt; no-op elsewhere
    __syncthreads();        // publish BC[cur] to all 4 waves
    if (l + 1 < LL) issue(l + 1, cur ^ 1);
    const size_t r = ((size_t)b << 9) + l;
    const float dt = delta[r * DI + d];
    const float xv = xa[r * DI + d];
    const float du = dt * xv;
    float acc = 0.f;
#pragma unroll
    for (int n = 0; n < NS; ++n) {
      h[n] = __expf(dt * A[n]) * h[n] + du * BC[cur][n];
      acc += h[n] * BC[cur][NS + n];
    }
    y[r * DI + d] = acc + dpd * xv;
    cur ^= 1;
  }
}

// ----------------------- gating: y * silu(z) -> bf16 ------------------------
__global__ void gate_kernel(const float* __restrict__ y, const float* __restrict__ z,
                            bf16_t* __restrict__ yg) {
  size_t i = (size_t)blockIdx.x * 256 + threadIdx.x;
  float zv = z[i];
  yg[i] = (bf16_t)(y[i] * zv / (1.0f + __expf(-zv)));
}

// ----------------------- GEMM3: yg(8192x512)@out_w(512x256)+res -------------
// grid 512, block 256 (8 waves x 2 tiles). A panel (16x512, 16KB) async-staged
// to LDS. Writes bf16 into concat half `colofs`; flip maps row back.
__global__ void gemm_out_kernel(const bf16_t* __restrict__ yg,
                                const bf16_t* __restrict__ wpak,
                                const float* __restrict__ x,
                                bf16_t* __restrict__ cat, int flip, int colofs) {
  __shared__ __align__(16) bf16_t As[16 * DI];   // 16KB
  const int tid  = threadIdx.x;
  const int lane = tid & 31;
  const int wave = tid >> 5;
  const int mt   = blockIdx.x;
  const int ntb  = wave * 2;
  const int KT   = DI / 32;                      // 16

  const bf16_t* asrc = yg + (size_t)mt * 16 * DI;
#pragma unroll
  for (int rnd = 0; rnd < 4; ++rnd) {            // 4 * 256 thr * 16B = 16KB
    int byteOff = (rnd * 256 + tid) * 16;
    async_copy_b128((const char*)asrc + byteOff, (char*)As + byteOff);
  }
  wait_async0();
  __syncthreads();

  v8f a0 = {}, a1 = {};
  for (int tk = 0; tk < KT; ++tk) {
    if (tk + 1 < KT)
      __builtin_prefetch(wpak + (((size_t)ntb * KT + tk + 1) * 32 + lane) * 16, 0, 1);
    v16bf a  = load_a_frag(As + tk * 32, DI, lane);
    v16bf b0 = load_b_frag(wpak, ntb + 0, tk, KT, lane);
    v16bf b1 = load_b_frag(wpak, ntb + 1, tk, KT, lane);
    a0 = wmma_bf16(a, b0, a0);
    a1 = wmma_bf16(a, b1, a1);
  }
  const int c0 = ntb * 16 + (lane & 15);
  const int rb = mt * 16 + ((lane >> 4) << 3);
  v8f acc[2] = {a0, a1};
#pragma unroll
  for (int j = 0; j < 2; ++j) {
    int c = c0 + j * 16;
#pragma unroll
    for (int i = 0; i < 8; ++i) {
      int r  = rb + i;                          // flipped-domain row
      int bb = r >> 9, l = r & (LL - 1);
      int sl = flip ? (LL - 1 - l) : l;
      size_t sr = ((size_t)bb << 9) + sl;       // original-domain row
      float v = acc[j][i] + x[sr * DM + c];     // residual
      cat[sr * (2 * DM) + colofs + c] = (bf16_t)v;
    }
  }
}

// ----------------------- proj GEMM: cat(8192x512)@proj_w + bias, ReLU -------
__global__ void gemm_proj_kernel(const bf16_t* __restrict__ cat,
                                 const bf16_t* __restrict__ wpak,
                                 const float* __restrict__ bias,
                                 float* __restrict__ yp) {
  __shared__ __align__(16) bf16_t As[16 * (2 * DM)];   // 16KB
  const int tid  = threadIdx.x;
  const int lane = tid & 31;
  const int wave = tid >> 5;
  const int mt   = blockIdx.x;
  const int ntb  = wave * 2;
  const int KT   = (2 * DM) / 32;               // 16

  const bf16_t* asrc = cat + (size_t)mt * 16 * (2 * DM);
#pragma unroll
  for (int rnd = 0; rnd < 4; ++rnd) {
    int byteOff = (rnd * 256 + tid) * 16;
    async_copy_b128((const char*)asrc + byteOff, (char*)As + byteOff);
  }
  wait_async0();
  __syncthreads();

  v8f a0 = {}, a1 = {};
  for (int tk = 0; tk < KT; ++tk) {
    v16bf a  = load_a_frag(As + tk * 32, 2 * DM, lane);
    v16bf b0 = load_b_frag(wpak, ntb + 0, tk, KT, lane);
    v16bf b1 = load_b_frag(wpak, ntb + 1, tk, KT, lane);
    a0 = wmma_bf16(a, b0, a0);
    a1 = wmma_bf16(a, b1, a1);
  }
  const int c0 = ntb * 16 + (lane & 15);
  const int rb = mt * 16 + ((lane >> 4) << 3);
  v8f acc[2] = {a0, a1};
#pragma unroll
  for (int j = 0; j < 2; ++j) {
    int c = c0 + j * 16;
#pragma unroll
    for (int i = 0; i < 8; ++i) {
      float v = acc[j][i] + bias[c];
      yp[(size_t)(rb + i) * DM + c] = v > 0.f ? v : 0.f;
    }
  }
}

// ----------------------- residual add + LayerNorm ---------------------------
__global__ void final_ln_kernel(const float* __restrict__ x, const float* __restrict__ yp,
                                const float* __restrict__ g, const float* __restrict__ beta,
                                float* __restrict__ out) {
  __shared__ float s1[DM], s2[DM];
  const int r = blockIdx.x, t = threadIdx.x;
  float v = x[(size_t)r * DM + t] + yp[(size_t)r * DM + t];
  s1[t] = v;
  s2[t] = v * v;
  __syncthreads();
  for (int s = 128; s > 0; s >>= 1) {
    if (t < s) { s1[t] += s1[t + s]; s2[t] += s2[t + s]; }
    __syncthreads();
  }
  float mu  = s1[0] * (1.0f / DM);
  float var = s2[0] * (1.0f / DM) - mu * mu;
  out[(size_t)r * DM + t] = (v - mu) * rsqrtf(var + 1e-5f) * g[t] + beta[t];
}

// ---------------------------------------------------------------------------
extern "C" void kernel_launch(void* const* d_in, const int* in_sizes, int n_in,
                              void* d_out, int out_size, void* d_ws, size_t ws_size,
                              hipStream_t stream) {
  (void)in_sizes; (void)n_in; (void)out_size; (void)ws_size;
  const float* x = (const float*)d_in[0];
  struct DirP {
    const float *in_w, *convw, *convb, *xproj, *dtw, *dtb, *Alog, *Dp, *out_w, *norm;
  } P[2];
  for (int dd = 0; dd < 2; ++dd) {
    int o = 1 + dd * 10;
    P[dd].in_w  = (const float*)d_in[o + 0];
    P[dd].convw = (const float*)d_in[o + 1];
    P[dd].convb = (const float*)d_in[o + 2];
    P[dd].xproj = (const float*)d_in[o + 3];
    P[dd].dtw   = (const float*)d_in[o + 4];
    P[dd].dtb   = (const float*)d_in[o + 5];
    P[dd].Alog  = (const float*)d_in[o + 6];
    P[dd].Dp    = (const float*)d_in[o + 7];
    P[dd].out_w = (const float*)d_in[o + 8];
    P[dd].norm  = (const float*)d_in[o + 9];
  }
  const float* proj_w = (const float*)d_in[21];
  const float* proj_b = (const float*)d_in[22];
  const float* ln_g   = (const float*)d_in[23];
  const float* ln_b   = (const float*)d_in[24];
  float* out = (float*)d_out;

  // bump allocator on d_ws (256B aligned chunks)
  char* wsb = (char*)d_ws;
  size_t off = 0;
  auto alloc = [&](size_t bytes) -> void* {
    void* p = wsb + off;
    off = (off + bytes + 255) & ~(size_t)255;
    return p;
  };
  bf16_t* xn    = (bf16_t*)alloc((size_t)RTOT * DM * 2);
  bf16_t* pIn   = (bf16_t*)alloc((size_t)DM * 2 * DI * 2);
  bf16_t* pOut  = (bf16_t*)alloc((size_t)DI * DM * 2);
  bf16_t* pProj = (bf16_t*)alloc((size_t)(2 * DM) * DM * 2);
  float*  xm    = (float*)alloc((size_t)RTOT * DI * 4);   // reused as y after scan
  float*  zb    = (float*)alloc((size_t)RTOT * DI * 4);
  float*  xa    = (float*)alloc((size_t)RTOT * DI * 4);
  float*  dl    = (float*)alloc((size_t)RTOT * DI * 4);
  float*  Bmb   = (float*)alloc((size_t)RTOT * NS * 4);
  float*  Cmb   = (float*)alloc((size_t)RTOT * NS * 4);
  bf16_t* yg    = (bf16_t*)alloc((size_t)RTOT * DI * 2);
  bf16_t* cat   = (bf16_t*)alloc((size_t)RTOT * 2 * DM * 2);
  float*  yproj = (float*)alloc((size_t)RTOT * DM * 4);

  const int ew = (RTOT * DI) / 256;   // elementwise grid

  pack_w_kernel<<<(2 * DM * DM) / 256, 256, 0, stream>>>(proj_w, pProj, 2 * DM, DM);

  for (int dd = 0; dd < 2; ++dd) {
    const int flip = dd;  // dd==1 is the flipped (backward) branch
    pack_w_kernel<<<(DM * 2 * DI) / 256, 256, 0, stream>>>(P[dd].in_w, pIn, DM, 2 * DI);
    pack_w_kernel<<<(DI * DM) / 256, 256, 0, stream>>>(P[dd].out_w, pOut, DI, DM);

    rmsnorm_kernel<<<RTOT, DM, 0, stream>>>(x, P[dd].norm, xn, flip);

    dim3 g1(RTOT / 16, 2);
    gemm_xz_kernel<<<g1, 256, 0, stream>>>(xn, pIn, xm, zb);

    conv_silu_kernel<<<ew, 256, 0, stream>>>(xm, P[dd].convw, P[dd].convb, xa);

    dbc_delta_kernel<<<RTOT, 256, 0, stream>>>(xa, P[dd].xproj, P[dd].dtw, P[dd].dtb,
                                               dl, Bmb, Cmb);

    scan_kernel<<<BB * (DI / 128), 128, 0, stream>>>(dl, xa, Bmb, Cmb,
                                                     P[dd].Alog, P[dd].Dp, xm);

    gate_kernel<<<ew, 256, 0, stream>>>(xm, zb, yg);

    gemm_out_kernel<<<RTOT / 16, 256, 0, stream>>>(yg, pOut, x, cat, flip, dd * DM);
  }

  gemm_proj_kernel<<<RTOT / 16, 256, 0, stream>>>(cat, pProj, proj_b, yproj);
  final_ln_kernel<<<RTOT, DM, 0, stream>>>(x, yproj, ln_g, ln_b, out);
}
